// SpanEncoder_14585708937887
// MI455X (gfx1250) — matrix operand order repaired and checked by reading, not compiled, not run on hardware
//
#include <hip/hip_runtime.h>
#include <hip/hip_bf16.h>

// ---------------------------------------------------------------------------
// SpanEncoder for MI455X (gfx1250, wave32).
//  Kernel 0: prep — transpose + zero-pad weights to f16 (removes all bounds
//            checks from the WMMA fragment path).
//  Kernel 1: per-token score MLP via v_wmma_f32_16x16x32_f16 (f32 accum).
//  Kernel 2: bandwidth-bound span writer; one wave32 per window; b128
//            NON-TEMPORAL stores (output is write-once, keep L2 for gathers).
// ---------------------------------------------------------------------------

#define L_TOK 8192
#define E_DIM 300
#define S_DIM 400
#define H_DIM 150
#define H_PAD 160          // 10 tiles of 16
#define K1PAD 416          // 13 tiles of 32 covering S_DIM=400
#define NT    10
#define MAX_N 10

typedef __attribute__((ext_vector_type(16))) _Float16 v16h;
typedef __attribute__((ext_vector_type(8)))  _Float16 v8h;
typedef __attribute__((ext_vector_type(8)))  float    v8f;
typedef __attribute__((ext_vector_type(4)))  float    f4;

// ws layout (bytes): [scores f32 x8192 | W1t f16 160x416 | W2t f16 160x160]
#define WS_W1T_OFF 32768
#define WS_W2T_OFF (32768 + H_PAD * K1PAD * 2)

// ---------------------------------------------------------------------------
// Kernel 0: W1 [400x150] -> W1t [160x416] f16 (transposed, zero-padded)
//           W2 [150x150] -> W2t [160x160] f16 (transposed, zero-padded)
// ---------------------------------------------------------------------------
__global__ __launch_bounds__(256)
void spanenc_prep_kernel(const float* __restrict__ W1,
                         const float* __restrict__ W2,
                         _Float16* __restrict__ W1t,
                         _Float16* __restrict__ W2t)
{
    const int idx = blockIdx.x * blockDim.x + threadIdx.x;
    if (idx < H_PAD * K1PAD) {
        const int n = idx / K1PAD, k = idx % K1PAD;
        const float v = (n < H_DIM && k < S_DIM) ? W1[(size_t)k * H_DIM + n] : 0.0f;
        W1t[idx] = (_Float16)v;
    } else {
        const int j = idx - H_PAD * K1PAD;
        if (j < H_PAD * H_PAD) {
            const int n = j / H_PAD, k = j % H_PAD;
            const float v = (n < H_DIM && k < H_DIM) ? W2[(size_t)k * H_DIM + n] : 0.0f;
            W2t[j] = (_Float16)v;
        }
    }
}

// ---------------------------------------------------------------------------
// Kernel 1: scores[L] = MLP(states).  128 threads = 4 waves; each wave owns
// 16 rows x all 10 column tiles.  Block covers 64 rows -> 128 blocks.
// Fragment lane maps per ISA 7.12.2 (wave32, 16-bit operands):
//   A: M = lane&15, K in {kb..kb+7} u {kb+16..kb+23}, kb = lane<16 ? 0 : 8
//   B: N = lane&15, K in {kb..kb+7} u {kb+16..kb+23}, kb = lane<16 ? 0 : 8
//   C: M = v + (lane<16 ? 0 : 8), N = lane&15
// ---------------------------------------------------------------------------
__global__ __launch_bounds__(128)
void spanenc_scores_kernel(const float* __restrict__ states,
                           const _Float16* __restrict__ W1t,
                           const float* __restrict__ b1,
                           const _Float16* __restrict__ W2t,
                           const float* __restrict__ b2,
                           const float* __restrict__ W3,
                           const float* __restrict__ b3,
                           float* __restrict__ scores)
{
    __shared__ _Float16 h1[64 * H_PAD];   // 20 KB
    __shared__ _Float16 h2[64 * H_PAD];   // 20 KB

    const int lane = threadIdx.x & 31;
    const int wave = threadIdx.x >> 5;
    const int rowBase = blockIdx.x * 64 + wave * 16;

    const int aRow = lane & 15;
    const int aKb  = (lane < 16) ? 0 : 8;
    const int bN   = lane & 15;
    const int bKb  = (lane < 16) ? 0 : 8;
    const int cM   = (lane < 16) ? 0 : 8;
    const int cN   = lane & 15;

    const v8f vzero = {};
    v8f acc[NT];
#pragma unroll
    for (int t = 0; t < NT; ++t) acc[t] = vzero;

    // ---------------- Layer 1: [16 x 400] x [400 x 160] ----------------
    const float* srow = states + (size_t)(rowBase + aRow) * S_DIM;

    // 12 full k-tiles, no bounds checks anywhere
    for (int kt = 0; kt < 12; ++kt) {
        const int k0 = kt * 32;
        const float* p = srow + k0 + aKb;
        const f4 f0 = *(const f4*)(p);
        const f4 f1 = *(const f4*)(p + 4);
        const f4 f2 = *(const f4*)(p + 16);
        const f4 f3 = *(const f4*)(p + 20);
        v16h a;
        a[0]=(_Float16)f0.x; a[1]=(_Float16)f0.y; a[2]=(_Float16)f0.z; a[3]=(_Float16)f0.w;
        a[4]=(_Float16)f1.x; a[5]=(_Float16)f1.y; a[6]=(_Float16)f1.z; a[7]=(_Float16)f1.w;
        a[8]=(_Float16)f2.x; a[9]=(_Float16)f2.y; a[10]=(_Float16)f2.z; a[11]=(_Float16)f2.w;
        a[12]=(_Float16)f3.x; a[13]=(_Float16)f3.y; a[14]=(_Float16)f3.z; a[15]=(_Float16)f3.w;
#pragma unroll
        for (int nt = 0; nt < NT; ++nt) {
            const int n = nt * 16 + bN;
            const v8h blo = *(const v8h*)(W1t + (size_t)n * K1PAD + k0 + bKb);
            const v8h bhi = *(const v8h*)(W1t + (size_t)n * K1PAD + k0 + bKb + 16);
            v16h b;
#pragma unroll
            for (int j = 0; j < 8; ++j) { b[j] = blo[j]; b[8 + j] = bhi[j]; }
            acc[nt] = __builtin_amdgcn_wmma_f32_16x16x32_f16(
                          false, a, false, b, (short)0, acc[nt], false, false);
        }
    }

    // tail k-tile (k0=384): each lane's first 8 K are valid (384..399),
    // second 8 are past the end -> compile-time zero fill, still no branches.
    {
        const int k0 = 384;
        const float* p = srow + k0 + aKb;
        const f4 f0 = *(const f4*)(p);
        const f4 f1 = *(const f4*)(p + 4);
        v16h a;
        a[0]=(_Float16)f0.x; a[1]=(_Float16)f0.y; a[2]=(_Float16)f0.z; a[3]=(_Float16)f0.w;
        a[4]=(_Float16)f1.x; a[5]=(_Float16)f1.y; a[6]=(_Float16)f1.z; a[7]=(_Float16)f1.w;
#pragma unroll
        for (int j = 8; j < 16; ++j) a[j] = (_Float16)0.0f;
#pragma unroll
        for (int nt = 0; nt < NT; ++nt) {
            const int n = nt * 16 + bN;
            const v8h blo = *(const v8h*)(W1t + (size_t)n * K1PAD + k0 + bKb);
            const v8h bhi = *(const v8h*)(W1t + (size_t)n * K1PAD + k0 + bKb + 16);
            v16h b;
#pragma unroll
            for (int j = 0; j < 8; ++j) { b[j] = blo[j]; b[8 + j] = bhi[j]; }
            acc[nt] = __builtin_amdgcn_wmma_f32_16x16x32_f16(
                          false, a, false, b, (short)0, acc[nt], false, false);
        }
    }

    // bias + ReLU -> h1 (f16; pad columns produce relu(0)=0 automatically,
    // because W1t pad columns are all-zero and bias is masked by n<H_DIM)
#pragma unroll
    for (int nt = 0; nt < NT; ++nt) {
        const int n = nt * 16 + cN;
        const float bias = (n < H_DIM) ? b1[n] : 0.0f;
#pragma unroll
        for (int v = 0; v < 8; ++v) {
            const float x = fmaxf(acc[nt][v] + bias, 0.0f);
            h1[(wave * 16 + cM + v) * H_PAD + n] = (_Float16)x;
        }
    }
    __syncthreads();

    // ---------------- Layer 2: [16 x 160] x [160 x 160] ----------------
#pragma unroll
    for (int t = 0; t < NT; ++t) acc[t] = vzero;

    const _Float16* hrow = h1 + (size_t)(wave * 16 + aRow) * H_PAD;
#pragma unroll
    for (int kt = 0; kt < 5; ++kt) {
        const int k0 = kt * 32;
        const v8h lo = *(const v8h*)(hrow + k0 + aKb);        // ds_load_b128
        const v8h hi = *(const v8h*)(hrow + k0 + aKb + 16);   // ds_load_b128
        v16h a;
#pragma unroll
        for (int j = 0; j < 8; ++j) { a[j] = lo[j]; a[8 + j] = hi[j]; }
#pragma unroll
        for (int nt = 0; nt < NT; ++nt) {
            const int n = nt * 16 + bN;
            const v8h blo = *(const v8h*)(W2t + (size_t)n * H_PAD + k0 + bKb);
            const v8h bhi = *(const v8h*)(W2t + (size_t)n * H_PAD + k0 + bKb + 16);
            v16h b;
#pragma unroll
            for (int j = 0; j < 8; ++j) { b[j] = blo[j]; b[8 + j] = bhi[j]; }
            acc[nt] = __builtin_amdgcn_wmma_f32_16x16x32_f16(
                          false, a, false, b, (short)0, acc[nt], false, false);
        }
    }

    // bias + ReLU -> h2
#pragma unroll
    for (int nt = 0; nt < NT; ++nt) {
        const int n = nt * 16 + cN;
        const float bias = (n < H_DIM) ? b2[n] : 0.0f;
#pragma unroll
        for (int v = 0; v < 8; ++v) {
            const float x = fmaxf(acc[nt][v] + bias, 0.0f);
            h2[(wave * 16 + cM + v) * H_PAD + n] = (_Float16)x;
        }
    }
    __syncthreads();

    // ---------------- Layer 3: [64 x 150] . [150] -> scores --------------
    if (threadIdx.x < 64) {
        const int r = threadIdx.x;
        const _Float16* row = h2 + (size_t)r * H_PAD;
        float s = b3[0];
        for (int h = 0; h < H_DIM; ++h)
            s += (float)row[h] * W3[h];
        scores[blockIdx.x * 64 + r] = s;
    }
}

// ---------------------------------------------------------------------------
// Kernel 2: one wave32 per window.  Row = [states[i] | states[i+n-1] | pooled]
// = 1100 floats = 275 f4 chunks (all segments 16B-aligned).  Output stores are
// non-temporal (write-once stream, 360 MB) so embeds/states stay L2-resident.
// ---------------------------------------------------------------------------
__global__ __launch_bounds__(256)
void spanenc_span_kernel(const float* __restrict__ embeds,
                         const float* __restrict__ states,
                         const float* __restrict__ scores,
                         float* __restrict__ out)
{
    const int lane  = threadIdx.x & 31;
    const int wid   = blockIdx.x * (blockDim.x >> 5) + (threadIdx.x >> 5);
    const int TOTAL = MAX_N * L_TOK - 45;
    if (wid >= TOTAL) return;

    // decode global window id -> (n, i); rows concatenated n=1..10
    int q = wid, n = 1;
    while (n < MAX_N && q >= (L_TOK - n + 1)) { q -= (L_TOK - n + 1); ++n; }
    const int i = q;

    // warm the n embed rows (global_prefetch) while we stream the state copies
    if (lane < n)
        __builtin_prefetch(embeds + (size_t)(i + lane) * E_DIM, 0, 3);

    // softmax over scores[i .. i+n-1], redundant per lane, all in registers
    float sc[MAX_N];
    float m = -3.0e38f;
#pragma unroll
    for (int k = 0; k < MAX_N; ++k)
        if (k < n) { sc[k] = scores[i + k]; m = fmaxf(m, sc[k]); }
    float den = 0.0f;
#pragma unroll
    for (int k = 0; k < MAX_N; ++k)
        if (k < n) { sc[k] = __expf(sc[k] - m); den += sc[k]; }
    const float inv = 1.0f / den;

    f4* __restrict__ orow = (f4*)(out + (size_t)wid * 1100);
    const f4* s0 = (const f4*)(states + (size_t)i * S_DIM);
    const f4* s1 = (const f4*)(states + (size_t)(i + n - 1) * S_DIM);

    // first-state / last-state segments: 100 f4 each, NT b128 stores
    for (int c = lane; c < 100; c += 32) {
        __builtin_nontemporal_store(s0[c], &orow[c]);
        __builtin_nontemporal_store(s1[c], &orow[100 + c]);
    }

    // pooled embeds: 75 f4 chunks, <=10 weighted b128 loads per chunk
    for (int c = lane; c < 75; c += 32) {
        f4 r = {};
#pragma unroll
        for (int k = 0; k < MAX_N; ++k) {
            if (k < n) {
                const f4 e = ((const f4*)(embeds + (size_t)(i + k) * E_DIM))[c];
                const float w = sc[k] * inv;
                r.x = fmaf(w, e.x, r.x);
                r.y = fmaf(w, e.y, r.y);
                r.z = fmaf(w, e.z, r.z);
                r.w = fmaf(w, e.w, r.w);
            }
        }
        __builtin_nontemporal_store(r, &orow[200 + c]);
    }
}

// ---------------------------------------------------------------------------
extern "C" void kernel_launch(void* const* d_in, const int* in_sizes, int n_in,
                              void* d_out, int out_size, void* d_ws, size_t ws_size,
                              hipStream_t stream) {
    const float* embeds = (const float*)d_in[0];
    const float* states = (const float*)d_in[1];
    const float* W1 = (const float*)d_in[2];
    const float* b1 = (const float*)d_in[3];
    const float* W2 = (const float*)d_in[4];
    const float* b2 = (const float*)d_in[5];
    const float* W3 = (const float*)d_in[6];
    const float* b3 = (const float*)d_in[7];
    float* out = (float*)d_out;

    float*    scores = (float*)d_ws;
    _Float16* W1t    = (_Float16*)((char*)d_ws + WS_W1T_OFF);
    _Float16* W2t    = (_Float16*)((char*)d_ws + WS_W2T_OFF);

    // Kernel 0: pad/transpose weights to f16 (92160 elements)
    const int prepElems  = H_PAD * K1PAD + H_PAD * H_PAD;
    const int prepBlocks = (prepElems + 255) / 256;
    spanenc_prep_kernel<<<prepBlocks, 256, 0, stream>>>(W1, W2, W1t, W2t);

    // Kernel 1: 128 blocks x 128 threads (4 waves of 16 rows each)
    spanenc_scores_kernel<<<L_TOK / 64, 128, 0, stream>>>(
        states, W1t, b1, W2t, b2, W3, b3, scores);

    // Kernel 2: one wave32 per window; 8 waves per 256-thread block
    const int total  = MAX_N * L_TOK - 45;           // 81875 windows
    const int blocks = (total + 7) / 8;
    spanenc_span_kernel<<<blocks, 256, 0, stream>>>(embeds, states, scores, out);
}
